// BinaryConv2dLayer_7035156431031
// MI455X (gfx1250) — matrix operand order repaired and checked by reading, not compiled, not run on hardware
//
#include <hip/hip_runtime.h>

typedef _Float16 v16h __attribute__((ext_vector_type(16)));
typedef _Float16 v8h  __attribute__((ext_vector_type(8)));
typedef _Float16 v2h  __attribute__((ext_vector_type(2)));
typedef float    v8f  __attribute__((ext_vector_type(8)));

#define C_IN   128
#define H_IN   56
#define W_IN   56
#define HW     (H_IN * W_IN)
#define OC     256
#define OH     54
#define OW     54
#define KTOT   1152          // 9 * 128, k = (kh*3+kw)*128 + ic
#define KSTEP  32
#define KITERS 36            // 1152 / 32
#define MTILE  64            // oc per block
#define MSUB   4             // four 16-row WMMA subtiles (M)
#define NSUB   2             // two 16-col WMMA subtiles (N) -> 32 pixels/wave
#define ROWB   (KTOT * 2 + 16)       // padded LDS row (2320 B): 580 dw, 580%64=4 -> banks spread
#define LDSW_BYTES (MTILE * ROWB)    // 148480 B (< 320 KB/WGP)

union U16H { v16h v; v2h p[8]; v8h h[2]; };

// builtin returns __fp16 vector; bit-identical to _Float16 vector
static __device__ inline v2h cvt_pk(float a, float b) {
    return __builtin_bit_cast(v2h, __builtin_amdgcn_cvt_pkrtz(a, b));
}

static __device__ inline v2h pk_sign2(float a, float b) {
    float sa = (a > 0.f) ? 1.f : ((a < 0.f) ? -1.f : 0.f);
    float sb = (b > 0.f) ? 1.f : ((b < 0.f) ? -1.f : 0.f);
    return cvt_pk(sa, sb);
}

__global__ __launch_bounds__(256)
void bwn_conv_wmma(const float* __restrict__ x,
                   const float* __restrict__ w,
                   float* __restrict__ out)
{
    __shared__ __align__(16) unsigned char ldsW[LDSW_BYTES];

    const int tid  = threadIdx.x;
    const int wave = tid >> 5;
    const int lane = tid & 31;

    const int oc_base = blockIdx.y * MTILE;

    // ---- cooperative, division-free fill: sign(w) -> f16 +-1 in LDS ----
    // 4 threads per oc row, each handles 288 consecutive k, packed 2-wide.
    {
        const int oc_f = tid >> 2;          // 0..63
        const int part = tid & 3;           // 0..3
        const float* wrow = w + (size_t)(oc_base + oc_f) * (C_IN * 9);
        unsigned char* lrow = ldsW + oc_f * ROWB;
        const int k0 = part * (KTOT / 4);   // 288, even; pairs never cross ic-128 boundary
        #pragma unroll 4
        for (int k = k0; k < k0 + KTOT / 4; k += 2) {
            int s  = k >> 7;                // kh*3+kw
            int ic = k & 127;
            float w0 = wrow[ic * 9 + s];
            float w1 = wrow[(ic + 1) * 9 + s];
            *(v2h*)(lrow + k * 2) = pk_sign2(w0, w1);
        }
    }
    __syncthreads();

    // ---- per-wave 64(M) x 32(N) output tile ----
    int slot = blockIdx.x * 8 + wave;       // (n, oh, ow-tile)
    int owt  = slot & 1;                    // 2 ow-tiles of 32
    int rest = slot >> 1;
    int oh   = rest % OH;
    int n    = rest / OH;

    const int nloc = lane & 15;             // N column / M row within 16x16 subtile
    const int hsel = (lane >> 4) & 1;       // half-wave select per ISA fragment layout
    const int owc0 = owt * 32 + nloc;       // column of B-fragment 0
    const int owc1 = owc0 + 16;             // column of B-fragment 1

    v8f acc[MSUB][NSUB];
    #pragma unroll
    for (int m = 0; m < MSUB; ++m)
        #pragma unroll
        for (int ns = 0; ns < NSUB; ++ns) acc[m][ns] = (v8f){};

    #pragma unroll 2
    for (int kb = 0; kb < KITERS; ++kb) {
        const int K0  = kb * KSTEP;
        const int s   = kb >> 2;            // (kh*3+kw)
        const int icb = (kb & 3) * 32;
        const int kh  = s / 3;
        const int kw  = s - 3 * kh;

        // B fragments: 32x16 activation tiles, packed f32->f16 converts
        const int ic0 = icb + hsel * 16;    // lanes 0-15: K0..15, lanes 16-31: K16..31
        const int ih  = oh + kh;
        int iw0 = owc0 + kw; iw0 = iw0 > (W_IN - 1) ? (W_IN - 1) : iw0;  // clamp; masked at store
        int iw1 = owc1 + kw; iw1 = iw1 > (W_IN - 1) ? (W_IN - 1) : iw1;
        const float* xb = x + ((size_t)(n * C_IN + ic0) * H_IN + ih) * W_IN;
        U16H b0, b1;
        #pragma unroll
        for (int t = 0; t < 8; ++t) {
            const float* p0 = xb + (2 * t) * HW;
            b0.p[t] = cvt_pk(p0[iw0], p0[HW + iw0]);
            b1.p[t] = cvt_pk(p0[iw1], p0[HW + iw1]);
        }

        // A fragments from LDS; each reused by NSUB WMMAs
        const int colOff = (K0 + hsel * 16) * 2;  // 16B aligned
        #pragma unroll
        for (int m = 0; m < MSUB; ++m) {
            const unsigned char* row = ldsW + (m * 16 + nloc) * ROWB + colOff;
            U16H a;
            a.h[0] = *(const v8h*)(row);          // K0 + hsel*16 .. +7
            a.h[1] = *(const v8h*)(row + 16);     // K0 + hsel*16 + 8 .. +15
            acc[m][0] = __builtin_amdgcn_wmma_f32_16x16x32_f16(
                false, a.v, false, b0.v, (short)0, acc[m][0], false, false);
            acc[m][1] = __builtin_amdgcn_wmma_f32_16x16x32_f16(
                false, a.v, false, b1.v, (short)0, acc[m][1], false, false);
        }
    }

    // ---- store 64x32 tile, non-temporal (keep x resident in L2); mask OW tail ----
    #pragma unroll
    for (int ns = 0; ns < NSUB; ++ns) {
        const int ow = owt * 32 + nloc + ns * 16;
        if (ow < OW) {
            #pragma unroll
            for (int m = 0; m < MSUB; ++m) {
                #pragma unroll
                for (int j = 0; j < 8; ++j) {
                    int oc = oc_base + m * 16 + j + hsel * 8;  // C layout: VGPR j -> rows j, j+8
                    __builtin_nontemporal_store(
                        acc[m][ns][j], &out[((size_t)(n * OC + oc) * OH + oh) * OW + ow]);
                }
            }
        }
    }
}

extern "C" void kernel_launch(void* const* d_in, const int* in_sizes, int n_in,
                              void* d_out, int out_size, void* d_ws, size_t ws_size,
                              hipStream_t stream) {
    const float* x = (const float*)d_in[0];
    const float* w = (const float*)d_in[1];
    float* out     = (float*)d_out;

    // 3456 wave-slots (32 n * 54 oh * 2 ow-tiles), 8 waves/block -> 432 blocks in x
    dim3 grid(432, OC / MTILE);   // (432, 4)
    dim3 block(256);
    bwn_conv_wmma<<<grid, block, 0, stream>>>(x, w, out);
}
